// MemoryTGNN_68367289418325
// MI455X (gfx1250) — matrix operand order repaired and checked by reading, not compiled, not run on hardware
//
#include <hip/hip_runtime.h>
#include <hip/hip_bf16.h>
#include <math.h>

// ---------------------------------------------------------------------------
// MemoryTGNN on gfx1250: WMMA f16->f32 GEMM pipeline with algebraic
// simplifications (zero-state GRUs, edge GEMM factored through nodes,
// decoder GEMM split into per-node zA/zB + per-edge emem@(A+B)).
// This revision interleaves two independent WMMA accumulator chains per wave
// where possible, hiding the F16 WMMA->WMMA hazard slots (ISA 7.12.1: 5 NOPs)
// and sharing each A fragment between two column tiles (half the ds_load_b128
// traffic for A in those stages).
// ---------------------------------------------------------------------------

typedef _Float16 v16h __attribute__((ext_vector_type(16)));
typedef _Float16 v8h  __attribute__((ext_vector_type(8)));
typedef float    v8f  __attribute__((ext_vector_type(8)));

#define H 128
#define H3 384

// ---- WMMA helpers ----------------------------------------------------------
// A fragment (16x32 f16) from an LDS row-major f16 tile [16][lda]:
// lane<16 : row=lane,   elems 0-15 = K kt*32+0..15
// lane>=16: row=lane-16, elems 0-15 = K kt*32+16..31
__device__ __forceinline__ v16h load_afrag(const _Float16* A, int lda, int kt,
                                           int r, int hi) {
  const _Float16* p = A + r * lda + kt * 32 + hi * 16;
  v8h lo = *(const v8h*)(p);
  v8h h8 = *(const v8h*)(p + 8);
  v16h a;
#pragma unroll
  for (int i = 0; i < 8; ++i) { a[i] = lo[i]; a[8 + i] = h8[i]; }
  return a;
}

// Packed B layout: frag(kt,nt): 32 lanes x 16 contiguous halves:
// element j of lane L = B[kt*32 + (L>>4)*16 + j][nt*16 + (L&15)]
__device__ __forceinline__ v8f wmma_rowtile(const _Float16* As, int lda,
                                            const _Float16* Bp, int ntiles,
                                            int nt, int lane) {
  const int r = lane & 15, hi = lane >> 4;
  v8f d = {};
#pragma unroll
  for (int kt = 0; kt < 4; ++kt) {   // K = 128 = 4 * 32
    v16h a = load_afrag(As, lda, kt, r, hi);
    v16h b = *(const v16h*)(Bp + (((size_t)kt * ntiles + nt) * 32 + lane) * 16);
    d = __builtin_amdgcn_wmma_f32_16x16x32_f16(false, a, false, b, (short)0, d,
                                               false, false);
  }
  return d;
}

// Two column tiles per wave: one shared A fragment feeds two interleaved
// WMMA accumulator chains (hides WMMA->WMMA dependency slots).
__device__ __forceinline__ void wmma_rowtile_pair(
    const _Float16* As, int lda,
    const _Float16* Bp0, int ntiles0, int nt0,
    const _Float16* Bp1, int ntiles1, int nt1,
    int lane, v8f& d0, v8f& d1) {
  const int r = lane & 15, hi = lane >> 4;
#pragma unroll
  for (int kt = 0; kt < 4; ++kt) {
    v16h a = load_afrag(As, lda, kt, r, hi);
    v16h b0 =
        *(const v16h*)(Bp0 + (((size_t)kt * ntiles0 + nt0) * 32 + lane) * 16);
    v16h b1 =
        *(const v16h*)(Bp1 + (((size_t)kt * ntiles1 + nt1) * 32 + lane) * 16);
    d0 = __builtin_amdgcn_wmma_f32_16x16x32_f16(false, a, false, b0, (short)0,
                                                d0, false, false);
    d1 = __builtin_amdgcn_wmma_f32_16x16x32_f16(false, a, false, b1, (short)0,
                                                d1, false, false);
  }
}

__device__ __forceinline__ float sigm(float x) { return 1.f / (1.f + __expf(-x)); }

// ordered-int encoding for float atomicMax
__device__ __forceinline__ unsigned encf(float f) {
  int i = __float_as_int(f);
  return (unsigned)(i >= 0 ? (i | 0x80000000) : ~i);
}
__device__ __forceinline__ float decf(unsigned u) {
  int i = (u & 0x80000000u) ? (int)(u & 0x7FFFFFFFu) : ~(int)u;
  return __int_as_float(i);
}

// ---- weight packing into B-fragment order ----------------------------------
// mode 0: B[k][n] = src[k*Nc+n]
// mode 1: B[k][n] = src[n*K+k]          (for @W.T GEMMs)
// mode 2: B[k][n] = src[k*Nc+n] + src[(k+128)*Nc+n]   (W_d1 top+bottom sum)
__global__ void pack_b_f16(const float* __restrict__ src,
                           _Float16* __restrict__ dst, int K, int Nc, int mode) {
  int total = K * Nc;
  int ntiles = Nc >> 4;
  for (int idx = blockIdx.x * blockDim.x + threadIdx.x; idx < total;
       idx += gridDim.x * blockDim.x) {
    int j = idx & 15;
    int lane = (idx >> 4) & 31;
    int rest = idx >> 9;
    int nt = rest % ntiles;
    int kt = rest / ntiles;
    int k = kt * 32 + (lane >> 4) * 16 + j;
    int n = nt * 16 + (lane & 15);
    float v;
    if (mode == 0)      v = src[(size_t)k * Nc + n];
    else if (mode == 1) v = src[(size_t)n * K + k];
    else                v = src[(size_t)k * Nc + n] + src[(size_t)(k + 128) * Nc + n];
    dst[idx] = (_Float16)v;
  }
}

// c_e[j] = bih_e[j] + dot(b_et, Wih_e[j][:])   (constant part of edge gx)
__global__ void ce_kernel(const float* __restrict__ b_et,
                          const float* __restrict__ Wih_e,
                          const float* __restrict__ bih_e,
                          float* __restrict__ ce) {
  int j = blockIdx.x * blockDim.x + threadIdx.x;
  if (j >= H3) return;
  float acc = bih_e[j];
  for (int k = 0; k < H; ++k) acc += b_et[k] * Wih_e[(size_t)j * H + k];
  ce[j] = acc;
}

__global__ void zero_f32(float* __restrict__ p, size_t cnt) {
  size_t stride = (size_t)gridDim.x * blockDim.x;
  for (size_t i = (size_t)blockIdx.x * blockDim.x + threadIdx.x; i < cnt;
       i += stride) p[i] = 0.f;
}

// ---- fused per-node kernel --------------------------------------------------
// 16 nodes/block, 256 threads = 8 wave32.
// x -> h=relu(x@Wenc+b) -> mem=GRU0(h) -> u=mem@Wet -> v=u@WihE^T (global)
//                                   \-> xw=mem@Wgat (global) -> a_s/a_d (global)
__global__ void node_fused(const float* __restrict__ x,
                           const float* __restrict__ b_enc,
                           const float* __restrict__ bih_n,
                           const float* __restrict__ bhh_n,
                           const float* __restrict__ att_src,
                           const float* __restrict__ att_dst,
                           const _Float16* __restrict__ pWenc,
                           const _Float16* __restrict__ pWihn,
                           const _Float16* __restrict__ pWet,
                           const _Float16* __restrict__ pWihe,
                           const _Float16* __restrict__ pWgat,
                           float* __restrict__ vout, float* __restrict__ xwout,
                           float* __restrict__ asout, float* __restrict__ adout,
                           int N) {
  __shared__ __align__(16) _Float16 bufA[16 * H];
  __shared__ __align__(16) _Float16 bufB[16 * H];
  __shared__ float sg[16 * H3];
  const int tid = threadIdx.x, w = tid >> 5, lane = tid & 31;
  const int r = lane & 15, hi = lane >> 4;
  const int n0 = blockIdx.x * 16;

  // stage 0: x tile -> f16 LDS
  for (int i = tid; i < 16 * H; i += 256) {
    int row = n0 + (i >> 7); if (row >= N) row = N - 1;
    bufA[i] = (_Float16)x[(size_t)row * H + (i & 127)];
  }
  __syncthreads();

  // S1: h = relu(x @ W_enc + b_enc) -> bufB
  {
    v8f d = wmma_rowtile(bufA, H, pWenc, 8, w, lane);
    int col = w * 16 + r;
    float be = b_enc[col];
#pragma unroll
    for (int vi = 0; vi < 8; ++vi)
      bufB[(vi + hi * 8) * H + col] = (_Float16)fmaxf(d[vi] + be, 0.f);
  }
  __syncthreads();

  // S2: gx = h @ Wih_n^T + bih_n -> sg[16][384]  (tiles w, w+8 paired; w+16)
  {
    v8f d0 = {}, d1 = {};
    wmma_rowtile_pair(bufB, H, pWihn, 24, w, pWihn, 24, w + 8, lane, d0, d1);
    v8f d2 = wmma_rowtile(bufB, H, pWihn, 24, w + 16, lane);
    int c0 = w * 16 + r, c1 = (w + 8) * 16 + r, c2 = (w + 16) * 16 + r;
    float b0 = bih_n[c0], b1 = bih_n[c1], b2 = bih_n[c2];
#pragma unroll
    for (int vi = 0; vi < 8; ++vi) {
      int row = vi + hi * 8;
      sg[row * H3 + c0] = d0[vi] + b0;
      sg[row * H3 + c1] = d1[vi] + b1;
      sg[row * H3 + c2] = d2[vi] + b2;
    }
  }
  __syncthreads();

  // GRU with zero state: mem = (1-z)*tanh(gn + r*bhh_n_n) -> bufA (f16)
  for (int i = tid; i < 16 * H; i += 256) {
    int m = i >> 7, c = i & 127;
    float r_ = sigm(sg[m * H3 + c] + bhh_n[c]);
    float z_ = sigm(sg[m * H3 + H + c] + bhh_n[H + c]);
    float n_ = tanhf(sg[m * H3 + 2 * H + c] + r_ * bhh_n[2 * H + c]);
    bufA[i] = (_Float16)((1.f - z_) * n_);
  }
  __syncthreads();

  // S3: u = mem @ W_et -> bufB
  {
    v8f d = wmma_rowtile(bufA, H, pWet, 8, w, lane);
    int col = w * 16 + r;
#pragma unroll
    for (int vi = 0; vi < 8; ++vi)
      bufB[(vi + hi * 8) * H + col] = (_Float16)d[vi];
  }
  __syncthreads();

  // S4: v = u @ Wih_e^T -> global [N][384]  (tiles w, w+8 paired; w+16)
  {
    v8f d0 = {}, d1 = {};
    wmma_rowtile_pair(bufB, H, pWihe, 24, w, pWihe, 24, w + 8, lane, d0, d1);
    v8f d2 = wmma_rowtile(bufB, H, pWihe, 24, w + 16, lane);
    int c0 = w * 16 + r, c1 = (w + 8) * 16 + r, c2 = (w + 16) * 16 + r;
#pragma unroll
    for (int vi = 0; vi < 8; ++vi) {
      int row = n0 + vi + hi * 8;
      if (row < N) {
        float* vr = vout + (size_t)row * H3;
        vr[c0] = d0[vi];
        vr[c1] = d1[vi];
        vr[c2] = d2[vi];
      }
    }
  }

  // S5: xw = mem @ W_gat -> global + sg (for attention reduction)
  {
    v8f d = wmma_rowtile(bufA, H, pWgat, 8, w, lane);
    int col = w * 16 + r;
#pragma unroll
    for (int vi = 0; vi < 8; ++vi) {
      int row = vi + hi * 8;
      if (n0 + row < N) xwout[(size_t)(n0 + row) * H + col] = d[vi];
      sg[row * H + col] = d[vi];
    }
  }
  __syncthreads();

  // a_s / a_d : per (node, head) 32-wide dot with att vectors
  if (tid < 64) {
    int m = tid >> 2, hd = tid & 3;
    float ss = 0.f, dd = 0.f;
    for (int c2 = 0; c2 < 32; ++c2) {
      float xv = sg[m * H + hd * 32 + c2];
      ss += xv * att_src[hd * 32 + c2];
      dd += xv * att_dst[hd * 32 + c2];
    }
    if (n0 + m < N) {
      asout[(n0 + m) * 4 + hd] = ss;
      adout[(n0 + m) * 4 + hd] = dd;
    }
  }
}

// ---- GAT: logits + per-dst max ---------------------------------------------
__global__ void gat_logits(const int* __restrict__ eidx,
                           const float* __restrict__ a_s,
                           const float* __restrict__ a_d,
                           float* __restrict__ logits,
                           unsigned* __restrict__ menc, int E, int N) {
  int t = blockIdx.x * blockDim.x + threadIdx.x;
  if (t >= (E + N) * 4) return;
  int ent = t >> 2, hd = t & 3;
  int s, dn;
  if (ent < E) { s = eidx[ent]; dn = eidx[E + ent]; } else { s = dn = ent - E; }
  float l = a_s[s * 4 + hd] + a_d[dn * 4 + hd];
  l = l > 0.f ? l : 0.2f * l;                   // leaky_relu 0.2
  logits[t] = l;
  atomicMax(&menc[dn * 4 + hd], encf(l));
}

// ---- GAT: exp + weighted scatter-add ---------------------------------------
__global__ void gat_accum(const int* __restrict__ eidx,
                          const float* __restrict__ logits,
                          const unsigned* __restrict__ menc,
                          const float* __restrict__ xw,
                          float* __restrict__ outacc, float* __restrict__ den,
                          int E, int N) {
  int e = blockIdx.x, c = threadIdx.x, hd = c >> 5;
  int s, dn;
  if (e < E) { s = eidx[e]; dn = eidx[E + e]; } else { s = dn = e - E; }
  float ex = __expf(logits[e * 4 + hd] - decf(menc[dn * 4 + hd]));
  atomicAdd(&outacc[(size_t)dn * H + c], ex * xw[(size_t)s * H + c]);
  if ((c & 31) == 0) atomicAdd(&den[dn * 4 + hd], ex);
}

// ---- GAT: normalize, relu, L2 -> z (f16) -----------------------------------
__global__ void gat_norm(const float* __restrict__ outacc,
                         const float* __restrict__ den,
                         const float* __restrict__ b_gat,
                         _Float16* __restrict__ zf16) {
  __shared__ float red[H];
  int n = blockIdx.x, c = threadIdx.x, hd = c >> 5;
  float hv = fmaxf(outacc[(size_t)n * H + c] / den[n * 4 + hd] + b_gat[c], 0.f);
  red[c] = hv * hv;
  __syncthreads();
  for (int s = 64; s > 0; s >>= 1) {
    if (c < s) red[c] += red[c + s];
    __syncthreads();
  }
  float nrm = fmaxf(sqrtf(red[0]), 1e-12f);
  zf16[(size_t)n * H + c] = (_Float16)(hv / nrm);
}

// ---- zA = z@W_d1_top, zB = z@W_d1_bot (paired: shared A fragment) ----------
__global__ void zab_kernel(const _Float16* __restrict__ zf16,
                           const _Float16* __restrict__ pA,
                           const _Float16* __restrict__ pB,
                           float* __restrict__ zA, float* __restrict__ zB,
                           int N) {
  __shared__ __align__(16) _Float16 zt[16 * H];
  int tid = threadIdx.x, w = tid >> 5, lane = tid & 31;
  int r = lane & 15, hi = lane >> 4;
  int n0 = blockIdx.x * 16;
  for (int i = tid; i < 16 * H; i += 256) {
    int row = n0 + (i >> 7); if (row >= N) row = N - 1;
    zt[i] = zf16[(size_t)row * H + (i & 127)];
  }
  __syncthreads();
  v8f d0 = {}, d1 = {};
  wmma_rowtile_pair(zt, H, pA, 8, w, pB, 8, w, lane, d0, d1);
  int col = w * 16 + r;
#pragma unroll
  for (int vi = 0; vi < 8; ++vi) {
    int row = n0 + vi + hi * 8;
    if (row < N) {
      zA[(size_t)row * H + col] = d0[vi];
      zB[(size_t)row * H + col] = d1[vi];
    }
  }
}

// ---- fused edge kernel: edge GRU -> emem@(A+B) -> decoder -> sigmoid -------
__global__ void edge_decoder(const int* __restrict__ eidx,
                             const int* __restrict__ pidx,
                             const float* __restrict__ v,
                             const float* __restrict__ ce,
                             const float* __restrict__ bhh_e,
                             const _Float16* __restrict__ pWd1S,
                             const float* __restrict__ zA,
                             const float* __restrict__ zB,
                             const float* __restrict__ b_d1,
                             const float* __restrict__ W_d2,
                             const float* __restrict__ b_d2,
                             float* __restrict__ out, int E) {
  __shared__ __align__(16) _Float16 sem[16 * H];
  __shared__ float ssc[16];
  __shared__ int sps[16], spd[16];
  int tid = threadIdx.x;
  int e0 = blockIdx.x * 16;
  if (tid < 16) {
    int ec = e0 + tid < E ? e0 + tid : E - 1;
    ssc[tid] = 0.f;
    sps[tid] = pidx[ec];
    spd[tid] = pidx[E + ec];
  }
  // edge GRU (zero state), gx folded through per-node v:
  //   gx = 0.5*(v[s]+v[d]) + c_e ;  emem = (1-z)*tanh(gn + r*bhh_e_n)
  for (int i = tid; i < 16 * H; i += 256) {
    int m = i >> 7, c = i & 127;
    int ec = e0 + m < E ? e0 + m : E - 1;
    int s = eidx[ec], dn = eidx[E + ec];
    const float* vs = v + (size_t)s * H3;
    const float* vd = v + (size_t)dn * H3;
    float g0 = 0.5f * (vs[c] + vd[c]) + ce[c];
    float g1 = 0.5f * (vs[H + c] + vd[H + c]) + ce[H + c];
    float g2 = 0.5f * (vs[2 * H + c] + vd[2 * H + c]) + ce[2 * H + c];
    float r_ = sigm(g0 + bhh_e[c]);
    float z_ = sigm(g1 + bhh_e[H + c]);
    float n_ = tanhf(g2 + r_ * bhh_e[2 * H + c]);
    sem[i] = (_Float16)((1.f - z_) * n_);
  }
  __syncthreads();

  int w = tid >> 5, lane = tid & 31, r = lane & 15, hi = lane >> 4;
  v8f d = wmma_rowtile(sem, H, pWd1S, 8, w, lane);   // emem @ (A+B)
  int col = w * 16 + r;
  float wd2 = W_d2[col], bd1 = b_d1[col];
#pragma unroll
  for (int vi = 0; vi < 8; ++vi) {
    int m = vi + hi * 8;
    float hv = d[vi] + zA[(size_t)sps[m] * H + col] +
               zB[(size_t)spd[m] * H + col] + bd1;
    hv = fmaxf(hv, 0.f);
    atomicAdd(&ssc[m], hv * wd2);                    // hdec . W_d2
  }
  __syncthreads();
  if (tid < 16 && e0 + tid < E)
    out[e0 + tid] = 1.f / (1.f + __expf(-(ssc[tid] + b_d2[0])));
}

// ---------------------------------------------------------------------------
extern "C" void kernel_launch(void* const* d_in, const int* in_sizes, int n_in,
                              void* d_out, int out_size, void* d_ws,
                              size_t ws_size, hipStream_t stream) {
  (void)n_in; (void)ws_size; (void)out_size;
  const float* x      = (const float*)d_in[0];
  const int*   eidx   = (const int*)d_in[1];
  const int*   pidx   = (const int*)d_in[2];
  const float* W_enc  = (const float*)d_in[3];
  const float* b_enc  = (const float*)d_in[4];
  const float* Wih_n  = (const float*)d_in[5];
  const float* bih_n  = (const float*)d_in[7];
  const float* bhh_n  = (const float*)d_in[8];
  const float* W_et   = (const float*)d_in[9];
  const float* b_et   = (const float*)d_in[10];
  const float* Wih_e  = (const float*)d_in[11];
  const float* bih_e  = (const float*)d_in[13];
  const float* bhh_e  = (const float*)d_in[14];
  const float* W_gat  = (const float*)d_in[15];
  const float* att_s  = (const float*)d_in[16];
  const float* att_d  = (const float*)d_in[17];
  const float* b_gat  = (const float*)d_in[18];
  const float* W_d1   = (const float*)d_in[19];
  const float* b_d1   = (const float*)d_in[20];
  const float* W_d2   = (const float*)d_in[21];
  const float* b_d2   = (const float*)d_in[22];

  const int N = in_sizes[0] / H;     // 50000
  const int E = in_sizes[1] / 2;     // 600000

  char* base = (char*)d_ws;
  size_t off = 0;
  auto alloc = [&](size_t b) -> void* {
    void* p = base + off;
    off = (off + b + 255) & ~(size_t)255;
    return p;
  };
  _Float16* pWenc = (_Float16*)alloc((size_t)H * H * 2);
  _Float16* pWihn = (_Float16*)alloc((size_t)H * H3 * 2);
  _Float16* pWet  = (_Float16*)alloc((size_t)H * H * 2);
  _Float16* pWihe = (_Float16*)alloc((size_t)H * H3 * 2);
  _Float16* pWgat = (_Float16*)alloc((size_t)H * H * 2);
  _Float16* pWd1A = (_Float16*)alloc((size_t)H * H * 2);
  _Float16* pWd1B = (_Float16*)alloc((size_t)H * H * 2);
  _Float16* pWd1S = (_Float16*)alloc((size_t)H * H * 2);
  float* cebuf    = (float*)alloc(H3 * 4);
  float* vbuf     = (float*)alloc((size_t)N * H3 * 4);
  float* xwbuf    = (float*)alloc((size_t)N * H * 4);
  float* asbuf    = (float*)alloc((size_t)N * 4 * 4);
  float* adbuf    = (float*)alloc((size_t)N * 4 * 4);
  float* logits   = (float*)alloc((size_t)(E + N) * 4 * 4);
  char*  accblk   = (char*)alloc((size_t)N * (4 + 4 + H) * 4);
  unsigned* menc  = (unsigned*)accblk;
  float* den      = (float*)(accblk + (size_t)N * 16);
  float* outacc   = (float*)(accblk + (size_t)N * 32);
  _Float16* zf16  = (_Float16*)alloc((size_t)N * H * 2);
  float* zAbuf    = (float*)alloc((size_t)N * H * 4);
  float* zBbuf    = (float*)alloc((size_t)N * H * 4);

  // 1) pack weights into B-fragment f16 layout
  pack_b_f16<<<64, 256, 0, stream>>>(W_enc, pWenc, H, H, 0);
  pack_b_f16<<<192, 256, 0, stream>>>(Wih_n, pWihn, H, H3, 1);
  pack_b_f16<<<64, 256, 0, stream>>>(W_et, pWet, H, H, 0);
  pack_b_f16<<<192, 256, 0, stream>>>(Wih_e, pWihe, H, H3, 1);
  pack_b_f16<<<64, 256, 0, stream>>>(W_gat, pWgat, H, H, 0);
  pack_b_f16<<<64, 256, 0, stream>>>(W_d1, pWd1A, H, H, 0);
  pack_b_f16<<<64, 256, 0, stream>>>(W_d1 + (size_t)H * H, pWd1B, H, H, 0);
  pack_b_f16<<<64, 256, 0, stream>>>(W_d1, pWd1S, H, H, 2);

  // 2) c_e = bih_e + b_et @ Wih_e^T
  ce_kernel<<<2, 256, 0, stream>>>(b_et, Wih_e, bih_e, cebuf);

  // 3) zero menc/den/outacc
  zero_f32<<<4096, 256, 0, stream>>>((float*)accblk, (size_t)N * (4 + 4 + H));

  // 4) fused per-node pipeline (5 WMMA stages)
  node_fused<<<(N + 15) / 16, 256, 0, stream>>>(
      x, b_enc, bih_n, bhh_n, att_s, att_d, pWenc, pWihn, pWet, pWihe, pWgat,
      vbuf, xwbuf, asbuf, adbuf, N);

  // 5) GAT softmax over destinations (with self loops)
  int tot = (E + N) * 4;
  gat_logits<<<(tot + 255) / 256, 256, 0, stream>>>(eidx, asbuf, adbuf, logits,
                                                    menc, E, N);
  gat_accum<<<E + N, 128, 0, stream>>>(eidx, logits, menc, xwbuf, outacc, den,
                                       E, N);
  gat_norm<<<N, 128, 0, stream>>>(outacc, den, b_gat, zf16);

  // 6) per-node decoder precompute zA/zB
  zab_kernel<<<(N + 15) / 16, 256, 0, stream>>>(zf16, pWd1A, pWd1B, zAbuf,
                                                zBbuf, N);

  // 7) fused per-edge GRU + decoder + sigmoid
  edge_decoder<<<(E + 15) / 16, 256, 0, stream>>>(
      eidx, pidx, vbuf, cebuf, bhh_e, pWd1S, zAbuf, zBbuf, b_d1, W_d2, b_d2,
      (float*)d_out, E);
}
